// GraphIDS_28149215658378
// MI455X (gfx1250) — matrix operand order repaired and checked by reading, not compiled, not run on hardware
//
#include <hip/hip_runtime.h>
#include <hip/hip_bf16.h>

// ---------------- types ----------------
typedef __bf16 bf16_t;
typedef __attribute__((ext_vector_type(16))) __bf16 v16bf;
typedef __attribute__((ext_vector_type(8)))  __bf16 v8bf;
typedef __attribute__((ext_vector_type(4)))  __bf16 v4bf;
typedef __attribute__((ext_vector_type(8)))  float  v8f;

#define T_TOK  65536   // B*S tokens (= E edges)
#define DMODEL 128
#define NHEAD  4
#define HDIM   32
#define SEQ    512
#define FFDIM  2048

// LDS byte offset of a __shared__-derived pointer (low 32 bits of generic ptr)
__device__ inline unsigned lds_off(const void* p) {
    return (unsigned)(unsigned long long)p;
}
// async global->LDS copies (CDNA5 Tensor/async path, tracked by ASYNCcnt)
__device__ inline void async_b128(unsigned lds, const void* gptr) {
    asm volatile("global_load_async_to_lds_b128 %0, %1, off"
                 :: "v"(lds), "v"((unsigned long long)gptr) : "memory");
}
__device__ inline void async_b128_off32(unsigned lds, const void* gptr) {
    asm volatile("global_load_async_to_lds_b128 %0, %1, off offset:32"
                 :: "v"(lds), "v"((unsigned long long)gptr) : "memory");
}
__device__ inline void async_b64(unsigned lds, const void* gptr) {
    asm volatile("global_load_async_to_lds_b64 %0, %1, off"
                 :: "v"(lds), "v"((unsigned long long)gptr) : "memory");
}
__device__ inline void wait_async0() {
    asm volatile("s_wait_asynccnt 0" ::: "memory");
}

// ---------------- tiny utility kernels ----------------
__global__ void k_zero(float* p, long n) {
    long i = (long)blockIdx.x * blockDim.x + threadIdx.x;
    for (; i < n; i += (long)gridDim.x * blockDim.x) p[i] = 0.f;
}

__global__ void k_f32_to_bf16(const float* __restrict__ s, bf16_t* __restrict__ d, long n) {
    long i = (long)blockIdx.x * blockDim.x + threadIdx.x;
    for (; i < n; i += (long)gridDim.x * blockDim.x) d[i] = (bf16_t)s[i];
}

// ---------------- GNN front-end ----------------
__global__ void k_scatter(const float* __restrict__ eattr, const int* __restrict__ eidx,
                          float* __restrict__ sums, float* __restrict__ cnt, int E) {
    long total = (long)E * 64;
    long i = (long)blockIdx.x * blockDim.x + threadIdx.x;
    for (; i < total; i += (long)gridDim.x * blockDim.x) {
        int e = (int)(i >> 6), j = (int)(i & 63);
        int dst = eidx[E + e];                       // edge_index[1][e]
        atomicAdd(&sums[(long)dst * 64 + j], eattr[i]);
        if (j == 0) atomicAdd(&cnt[dst], 1.f);
    }
}

__global__ void k_sage(const float* __restrict__ sums, const float* __restrict__ cnt,
                       const float* __restrict__ nattr,
                       const float* __restrict__ wn, const float* __restrict__ bn,
                       const float* __restrict__ wx, const float* __restrict__ bx,
                       float* __restrict__ h, int N) {
    long total = (long)N * 128;
    long i = (long)blockIdx.x * blockDim.x + threadIdx.x;
    for (; i < total; i += (long)gridDim.x * blockDim.x) {
        int n = (int)(i >> 7), j = (int)(i & 127);
        float c = fmaxf(cnt[n], 1.f);
        float acc = bn[j] + bx[j];
        const float* sp = sums + (long)n * 64;
        for (int k = 0; k < 64; ++k)  acc += (sp[k] / c) * wn[k * 128 + j];
        const float* np = nattr + (long)n * 128;
        for (int k = 0; k < 128; ++k) acc += np[k] * wx[k * 128 + j];
        h[i] = fmaxf(acc, 0.f);
    }
}

__global__ void k_edge(const float* __restrict__ h, const int* __restrict__ cpl,
                       const float* __restrict__ we, const float* __restrict__ be,
                       float* __restrict__ e, int E) {
    long total = (long)E * 64;
    long i = (long)blockIdx.x * blockDim.x + threadIdx.x;
    for (; i < total; i += (long)gridDim.x * blockDim.x) {
        int ed = (int)(i >> 6), j = (int)(i & 63);
        int a = cpl[(long)ed * 2 + 0], b = cpl[(long)ed * 2 + 1];
        float acc = be[j];
        const float* ha = h + (long)a * 128;
        const float* hb = h + (long)b * 128;
        for (int k = 0; k < 128; ++k) acc += ha[k] * we[k * 64 + j];
        for (int k = 0; k < 128; ++k) acc += hb[k] * we[(128 + k) * 64 + j];
        e[i] = acc;
    }
}

// =====================================================================
// WMMA GEMM: block = 128(M) x 64(N) tile, 8 waves, each wave 16x64.
// A (bf16, row-major MxK), W (bf16, row-major KxN).
// Tiles staged via async global->LDS (ASYNCcnt), fragments via ds loads.
// =====================================================================
template <int RELU, int OUTBF>
__global__ void k_gemm(const bf16_t* __restrict__ A, const bf16_t* __restrict__ W,
                       const float* __restrict__ bias, void* __restrict__ Cout,
                       int M, int N, int K) {
    __shared__ bf16_t ash[128 * 32];   // [row][k]   8 KB
    __shared__ bf16_t bsh[32 * 64];    // [k][n]     4 KB

    int tiles_n = N >> 6;
    int bm = blockIdx.x / tiles_n;
    int bn = blockIdx.x - bm * tiles_n;
    int wv   = threadIdx.x >> 5;
    int lane = threadIdx.x & 31;
    int half = lane >> 4, lcol = lane & 15;
    int kadd = half * 8;

    int row0b = bm * 128;          // block row base
    int col0  = bn * 64;           // block col base
    v8f acc[4] = {};

    // per-thread staging coordinates (constant across K loop)
    int aidx = threadIdx.x * 16;               // element in [128 x 32]
    int ar = aidx >> 5, ac = aidx & 31;        // row, k-col (0 or 16)
    int bidx = threadIdx.x * 8;
    int kr = bidx >> 6, nc = bidx & 63;        // k-row, n-col (multiple of 8)
    unsigned lA = lds_off(ash + ar * 32 + ac);
    unsigned lB = lds_off(bsh + kr * 64 + nc);

    for (int kk = 0; kk < K; kk += 32) {
        __syncthreads();                       // previous tile fully consumed
        async_b128(lA, A + (long)(row0b + ar) * K + kk + ac);        // 16B
        async_b128(lA + 16, A + (long)(row0b + ar) * K + kk + ac + 8);  // next 16B
        async_b128(lB, W + (long)(kk + kr) * N + col0 + nc);
        wait_async0();
        __syncthreads();                       // tile visible to all waves

        // A fragment (16x32) for this wave from LDS
        v16bf a;
        {
            const v8bf* ap = (const v8bf*)(ash + (wv * 16 + lcol) * 32 + kadd);
            v8bf alo = ap[0], ahi = ap[2];
#pragma unroll
            for (int i = 0; i < 8; ++i) { a[i] = alo[i]; a[i + 8] = ahi[i]; }
        }
#pragma unroll
        for (int t = 0; t < 4; ++t) {
            v16bf b;
            int bc = t * 16 + lcol;
            int kb = half * 8;
#pragma unroll
            for (int i = 0; i < 16; ++i) b[i] = bsh[(kb + i + (i & 8)) * 64 + bc];
            acc[t] = __builtin_amdgcn_wmma_f32_16x16x32_bf16(false, a, false, b,
                                                             (short)0, acc[t], false, false);
        }
    }

    int rbase = row0b + wv * 16 + half * 8;
#pragma unroll
    for (int t = 0; t < 4; ++t) {
        int col = col0 + t * 16 + lcol;
        float bv = bias ? bias[col] : 0.f;
#pragma unroll
        for (int r = 0; r < 8; ++r) {
            float v = acc[t][r] + bv;
            if (RELU) v = fmaxf(v, 0.f);
            long idx = (long)(rbase + r) * N + col;
            if (OUTBF) ((bf16_t*)Cout)[idx] = (bf16_t)v;
            else       ((float*)Cout)[idx] = v;
        }
    }
}

// =====================================================================
// Flash attention per (batch, head). Q,K,V bf16 [T, DMODEL]; O bf16.
// S^T = Kchunk(16x32) @ Q^T(32x16): each lane owns 16 scores of ONE query
// -> softmax reductions are per-lane + one shfl_xor(16).
// K/V chunks staged via async global->LDS. grid = B*NHEAD*4, 256 thr.
// =====================================================================
__global__ void k_attn(const bf16_t* __restrict__ Q, const bf16_t* __restrict__ Kb,
                       const bf16_t* __restrict__ V, bf16_t* __restrict__ O) {
    __shared__ bf16_t ksh[32 * 32];                 // [key][dim] chunk  2 KB
    __shared__ bf16_t vsh[32 * 32];                 // [key][dim] chunk  2 KB
    __shared__ bf16_t psh[8][16 * 32];              // [wave][query][key] 8 KB
    __shared__ __align__(32) float csh[8][16];      // per-wave per-query factors

    int blk  = blockIdx.x;
    int qt8  = blk & 3;
    int bh   = blk >> 2;
    int h    = bh & (NHEAD - 1);
    int b    = bh >> 2;                 // NHEAD == 4
    int wv   = threadIdx.x >> 5;
    int lane = threadIdx.x & 31;
    int half = lane >> 4, lcol = lane & 15;
    int kadd = half * 8;
    int qrow0 = b * SEQ + (qt8 * 8 + wv) * 16;

    // Q^T B-fragment: element (k=dim, n=query) = Q[query, dim]; loaded once.
    v16bf qbf;
    {
        const v8bf* qp = (const v8bf*)(Q + (long)(qrow0 + lcol) * DMODEL + h * HDIM + half * 8);
        v8bf q0 = qp[0], q1 = qp[2];
#pragma unroll
        for (int i = 0; i < 8; ++i) { qbf[i] = q0[i]; qbf[i + 8] = q1[i]; }
    }

    v8f o0 = {}, o1 = {};
    float m_ln = -1e30f, l_ln = 0.f;    // running stats for query `lcol`
    const float scale = 0.17677669529663687f;   // 1/sqrt(32)

    // staging coordinates: 4 bf16 (8B) of K and of V per thread per chunk
    int sidx = threadIdx.x * 4;
    int skey = sidx >> 5, sdim = sidx & 31;
    unsigned lK = lds_off(ksh + skey * 32 + sdim);
    unsigned lV = lds_off(vsh + skey * 32 + sdim);

    for (int kc = 0; kc < SEQ; kc += 32) {
        __syncthreads();
        long g = (long)(b * SEQ + kc + skey) * DMODEL + h * HDIM + sdim;
        async_b64(lK, Kb + g);
        async_b64(lV, V + g);
        wait_async0();
        __syncthreads();

        // S^T tiles: A = keys x dims from LDS, B = Q^T
        v16bf ka0, ka1;
        {
            const v8bf* p0 = (const v8bf*)(ksh + lcol * 32 + kadd);          // keys 0-15
            const v8bf* p1 = (const v8bf*)(ksh + (16 + lcol) * 32 + kadd);   // keys 16-31
            v8bf a0 = p0[0], a1 = p0[2], b0 = p1[0], b1 = p1[2];
#pragma unroll
            for (int i = 0; i < 8; ++i) {
                ka0[i] = a0[i]; ka0[i + 8] = a1[i];
                ka1[i] = b0[i]; ka1[i + 8] = b1[i];
            }
        }
        v8f zc = {};
        v8f s0 = __builtin_amdgcn_wmma_f32_16x16x32_bf16(false, ka0, false, qbf, (short)0, zc, false, false);
        v8f s1 = __builtin_amdgcn_wmma_f32_16x16x32_bf16(false, ka1, false, qbf, (short)0, zc, false, false);

        // per-lane online softmax for query `lcol`
        float cmax = -1e30f;
#pragma unroll
        for (int r = 0; r < 8; ++r) {
            s0[r] *= scale; s1[r] *= scale;
            cmax = fmaxf(cmax, fmaxf(s0[r], s1[r]));
        }
        cmax = fmaxf(cmax, __shfl_xor(cmax, 16));   // merge the two key-halves
        float mn = fmaxf(m_ln, cmax);
        float cf = __expf(m_ln - mn);
        float rs = 0.f;
        v8bf p0w, p1w;
#pragma unroll
        for (int r = 0; r < 8; ++r) {
            float e0 = __expf(s0[r] - mn);
            float e1 = __expf(s1[r] - mn);
            rs += e0 + e1;
            p0w[r] = (bf16_t)e0;
            p1w[r] = (bf16_t)e1;
        }
        rs += __shfl_xor(rs, 16);
        l_ln = l_ln * cf + rs;
        m_ln = mn;

        // publish rescale factor + transposed P (query-major) to this wave's LDS
        if (half == 0) csh[wv][lcol] = cf;
        *(v8bf*)(psh[wv] + lcol * 32 + kadd)      = p0w;   // keys kadd..kadd+7
        *(v8bf*)(psh[wv] + lcol * 32 + 16 + kadd) = p1w;   // keys 16+kadd..
        __syncthreads();

        // rescale O by per-row factors (rows = queries half*8 .. half*8+7)
        v8f cr = *(const v8f*)(&csh[wv][half * 8]);
#pragma unroll
        for (int r = 0; r < 8; ++r) { o0[r] *= cr[r]; o1[r] *= cr[r]; }

        // P A-fragment (queries x keys) and V B-fragments (keys x dims)
        v16bf pa;
        {
            const v8bf* pp = (const v8bf*)(psh[wv] + lcol * 32 + kadd);
            v8bf plo = pp[0], phi = pp[2];
#pragma unroll
            for (int i = 0; i < 8; ++i) { pa[i] = plo[i]; pa[i + 8] = phi[i]; }
        }
        v16bf vb0, vb1;
        {
            int kb = half * 8;
#pragma unroll
            for (int i = 0; i < 16; ++i) {
                vb0[i] = vsh[(kb + i + (i & 8)) * 32 + lcol];
                vb1[i] = vsh[(kb + i + (i & 8)) * 32 + 16 + lcol];
            }
        }
        o0 = __builtin_amdgcn_wmma_f32_16x16x32_bf16(false, pa, false, vb0, (short)0, o0, false, false);
        o1 = __builtin_amdgcn_wmma_f32_16x16x32_bf16(false, pa, false, vb1, (short)0, o1, false, false);
    }

    // final normalization: bounce 1/l per query through LDS
    if (half == 0) csh[wv][lcol] = 1.f / l_ln;
    __syncthreads();
    v8f lr = *(const v8f*)(&csh[wv][half * 8]);
#pragma unroll
    for (int r = 0; r < 8; ++r) {
        long row = qrow0 + r + half * 8;
        O[row * DMODEL + h * HDIM + lcol]      = (bf16_t)(o0[r] * lr[r]);
        O[row * DMODEL + h * HDIM + 16 + lcol] = (bf16_t)(o1[r] * lr[r]);
    }
}

// ---------------- fused residual-add + LayerNorm, emits f32 + bf16 shadow ----------------
__global__ void k_add_ln(const float* __restrict__ x, const float* __restrict__ y,
                         const float* __restrict__ g, const float* __restrict__ bta,
                         float* __restrict__ out, bf16_t* __restrict__ outbf) {
    int row = blockIdx.x;
    int j = threadIdx.x;   // 128 threads
    __shared__ float red[128];
    float v = x[(long)row * 128 + j] + y[(long)row * 128 + j];
    red[j] = v; __syncthreads();
    for (int s = 64; s > 0; s >>= 1) { if (j < s) red[j] += red[j + s]; __syncthreads(); }
    float mu = red[0] / 128.f; __syncthreads();
    float d = v - mu;
    red[j] = d * d; __syncthreads();
    for (int s = 64; s > 0; s >>= 1) { if (j < s) red[j] += red[j + s]; __syncthreads(); }
    float var = red[0] / 128.f;
    float o = d * rsqrtf(var + 1e-5f) * g[j] + bta[j];
    out[(long)row * 128 + j] = o;
    outbf[(long)row * 128 + j] = (bf16_t)o;
}

// ---------------- host orchestration ----------------
// d_in flattening (setup_inputs dict insertion order, recursive):
enum {
    I_EIDX = 0, I_EATTR = 1, I_ECPL = 2, I_NN = 3, I_NATTR = 4,
    I_WNEIGH = 5, I_BNEIGH = 6, I_WNODE = 7, I_BNODE = 8, I_WEDGE = 9, I_BEDGE = 10,
    I_INW = 11, I_INB = 12,
    I_ENC0 = 13,   // stride 16: wq wk wv wo bq bk bv bo ln1g ln1b w1 b1 w2 b2 ln2g ln2b
    I_DEC0 = 45,   // stride 26: sa(8) ca(8) ln1g ln1b ln2g ln2b ln3g ln3b w1 b1 w2 b2
    I_OUTW = 97, I_OUTB = 98
};

static inline int nblk(long n, int tpb = 256, long cap = 262144) {
    long b = (n + tpb - 1) / tpb;
    if (b > cap) b = cap;
    return (int)b;
}

extern "C" void kernel_launch(void* const* d_in, const int* in_sizes, int n_in,
                              void* d_out, int out_size, void* d_ws, size_t ws_size,
                              hipStream_t stream) {
    const int E = in_sizes[I_EATTR] / 64;
    const int N = in_sizes[I_NATTR] / 128;
    const long T = T_TOK;

    // ---- workspace bump allocator ----
    size_t off = 0;
    auto alloc = [&](size_t bytes) -> void* {
        void* p = (char*)d_ws + off;
        off += (bytes + 255) & ~(size_t)255;
        return p;
    };
    float*  sums   = (float*)alloc((size_t)N * 64 * 4);
    float*  cnt    = (float*)alloc((size_t)N * 4);
    float*  hbuf   = (float*)alloc((size_t)N * 128 * 4);
    float*  ebuf   = (float*)alloc((size_t)E * 64 * 4);
    float*  src    = (float*)alloc((size_t)T * 128 * 4);
    float*  mem    = (float*)alloc((size_t)T * 128 * 4);
    float*  cur    = (float*)alloc((size_t)T * 128 * 4);
    float*  t2     = (float*)alloc((size_t)T * 128 * 4);
    float*  t3     = (float*)alloc((size_t)T * 128 * 4);
    bf16_t* src_bf = (bf16_t*)alloc((size_t)T * 128 * 2);
    bf16_t* mem_bf = (bf16_t*)alloc((size_t)T * 128 * 2);
    bf16_t* cur_bf = (bf16_t*)alloc((size_t)T * 128 * 2);
    bf16_t* t3_bf  = (bf16_t*)alloc((size_t)T * 128 * 2);
    bf16_t* qb     = (bf16_t*)alloc((size_t)T * 128 * 2);
    bf16_t* kb     = (bf16_t*)alloc((size_t)T * 128 * 2);
    bf16_t* vb     = (bf16_t*)alloc((size_t)T * 128 * 2);
    bf16_t* xb     = (bf16_t*)alloc((size_t)T * 128 * 2);   // attn out / misc bf16 input
    bf16_t* hid    = (bf16_t*)alloc((size_t)T * FFDIM * 2);

    // ---- convert weights once per call into bf16 pool ----
    auto cvtw = [&](int idx, long n) -> bf16_t* {
        bf16_t* p = (bf16_t*)alloc((size_t)n * 2);
        k_f32_to_bf16<<<nblk(n), 256, 0, stream>>>((const float*)d_in[idx], p, n);
        return p;
    };
    struct AttnW { bf16_t *wq, *wk, *wv, *wo; };
    struct EncW { AttnW a; bf16_t *w1, *w2; } encw[2];
    struct DecW { AttnW sa, ca; bf16_t *w1, *w2; } decw[2];
    bf16_t* w_in = cvtw(I_INW, 64 * 128);
    for (int lyr = 0; lyr < 2; ++lyr) {
        int base = I_ENC0 + 16 * lyr;
        encw[lyr].a.wq = cvtw(base + 0, 128 * 128);
        encw[lyr].a.wk = cvtw(base + 1, 128 * 128);
        encw[lyr].a.wv = cvtw(base + 2, 128 * 128);
        encw[lyr].a.wo = cvtw(base + 3, 128 * 128);
        encw[lyr].w1   = cvtw(base + 10, 128 * FFDIM);
        encw[lyr].w2   = cvtw(base + 12, FFDIM * 128);
    }
    for (int lyr = 0; lyr < 2; ++lyr) {
        int base = I_DEC0 + 26 * lyr;
        decw[lyr].sa.wq = cvtw(base + 0, 128 * 128);
        decw[lyr].sa.wk = cvtw(base + 1, 128 * 128);
        decw[lyr].sa.wv = cvtw(base + 2, 128 * 128);
        decw[lyr].sa.wo = cvtw(base + 3, 128 * 128);
        decw[lyr].ca.wq = cvtw(base + 8, 128 * 128);
        decw[lyr].ca.wk = cvtw(base + 9, 128 * 128);
        decw[lyr].ca.wv = cvtw(base + 10, 128 * 128);
        decw[lyr].ca.wo = cvtw(base + 11, 128 * 128);
        decw[lyr].w1    = cvtw(base + 22, 128 * FFDIM);
        decw[lyr].w2    = cvtw(base + 24, FFDIM * 128);
    }
    bf16_t* w_out = cvtw(I_OUTW, 128 * 64);

    // ---- launch helpers ----
    auto conv = [&](const float* s, bf16_t* d, long n) {
        k_f32_to_bf16<<<nblk(n), 256, 0, stream>>>(s, d, n);
    };
    auto gemm = [&](const bf16_t* A, const bf16_t* W, const float* bias, void* C,
                    int M, int Nn, int K, int relu, int outbf) {
        int blocks = (M / 128) * (Nn / 64);
        if (relu)       k_gemm<1, 1><<<blocks, 256, 0, stream>>>(A, W, bias, C, M, Nn, K);
        else if (outbf) k_gemm<0, 1><<<blocks, 256, 0, stream>>>(A, W, bias, C, M, Nn, K);
        else            k_gemm<0, 0><<<blocks, 256, 0, stream>>>(A, W, bias, C, M, Nn, K);
    };
    auto addln = [&](const float* x, const float* y, const float* g, const float* b,
                     float* o, bf16_t* obf) {
        k_add_ln<<<(int)T, 128, 0, stream>>>(x, y, g, b, o, obf);
    };
    auto run_attn = [&](const bf16_t* qin_bf, const bf16_t* kvin_bf, const AttnW& w,
                        const float* bq, const float* bkk, const float* bvv, const float* bo,
                        float* dest) {
        gemm(qin_bf,  w.wq, bq,  qb, (int)T, 128, 128, 0, 1);
        gemm(kvin_bf, w.wk, bkk, kb, (int)T, 128, 128, 0, 1);
        gemm(kvin_bf, w.wv, bvv, vb, (int)T, 128, 128, 0, 1);
        k_attn<<<128 * NHEAD * 4, 256, 0, stream>>>(qb, kb, vb, xb);
        gemm(xb, w.wo, bo, dest, (int)T, 128, 128, 0, 0);
    };
    auto run_ffn = [&](const bf16_t* xin_bf, const bf16_t* w1, const float* b1,
                       const bf16_t* w2, const float* b2, float* dest) {
        gemm(xin_bf, w1, b1, hid, (int)T, FFDIM, 128, 1, 1);     // relu + bf16 out
        gemm(hid, w2, b2, dest, (int)T, 128, FFDIM, 0, 0);
    };
    auto P = [&](int i) { return (const float*)d_in[i]; };

    // ---- GNN front-end ----
    k_zero<<<nblk((long)N * 64), 256, 0, stream>>>(sums, (long)N * 64);
    k_zero<<<nblk(N), 256, 0, stream>>>(cnt, N);
    k_scatter<<<nblk((long)E * 64), 256, 0, stream>>>(P(I_EATTR), (const int*)d_in[I_EIDX], sums, cnt, E);
    k_sage<<<nblk((long)N * 128), 256, 0, stream>>>(sums, cnt, P(I_NATTR),
                                                    P(I_WNEIGH), P(I_BNEIGH), P(I_WNODE), P(I_BNODE),
                                                    hbuf, N);
    k_edge<<<nblk((long)E * 64), 256, 0, stream>>>(hbuf, (const int*)d_in[I_ECPL],
                                                   P(I_WEDGE), P(I_BEDGE), ebuf, E);
    // input projection: src = e @ in_w + in_b  (and bf16 shadow)
    conv(ebuf, xb, (long)E * 64);
    gemm(xb, w_in, P(I_INB), src, (int)T, 128, 64, 0, 0);
    conv(src, src_bf, T * 128);

    // ---- encoder ----
    for (int lyr = 0; lyr < 2; ++lyr) {
        int base = I_ENC0 + 16 * lyr;
        const float*  xin   = (lyr == 0) ? src    : mem;
        const bf16_t* xinbf = (lyr == 0) ? src_bf : mem_bf;
        run_attn(xinbf, xinbf, encw[lyr].a, P(base + 4), P(base + 5), P(base + 6), P(base + 7), t2);
        addln(xin, t2, P(base + 8), P(base + 9), t3, t3_bf);
        run_ffn(t3_bf, encw[lyr].w1, P(base + 11), encw[lyr].w2, P(base + 13), t2);
        addln(t3, t2, P(base + 14), P(base + 15), mem, mem_bf);
    }

    // ---- decoder (cross-attending mem) ----
    for (int lyr = 0; lyr < 2; ++lyr) {
        int base = I_DEC0 + 26 * lyr;
        const float*  xin   = (lyr == 0) ? src    : cur;
        const bf16_t* xinbf = (lyr == 0) ? src_bf : cur_bf;
        run_attn(xinbf, xinbf, decw[lyr].sa, P(base + 4), P(base + 5), P(base + 6), P(base + 7), t2);
        addln(xin, t2, P(base + 16), P(base + 17), t3, t3_bf);
        run_attn(t3_bf, mem_bf, decw[lyr].ca, P(base + 12), P(base + 13), P(base + 14), P(base + 15), t2);
        addln(t3, t2, P(base + 18), P(base + 19), cur, cur_bf);
        run_ffn(cur_bf, decw[lyr].w1, P(base + 23), decw[lyr].w2, P(base + 25), t2);
        addln(cur, t2, P(base + 20), P(base + 21), cur, cur_bf);
    }

    // ---- output projection to d_out ----
    gemm(cur_bf, w_out, P(I_OUTB), d_out, (int)T, 64, 128, 0, 0);
}
